// SpikingHiddenLayer_7189775254184
// MI455X (gfx1250) — matrix operand order repaired and checked by reading, not compiled, not run on hardware
//
#include <hip/hip_runtime.h>

// ---------------------------------------------------------------------------
// SNN forward for MI455X (gfx1250, wave32, WMMA + TDM).
//   Phase 1: h1 = inputs @ w_input  — bf16 WMMA, A tiles streamed by the
//            Tensor Data Mover (double-buffered, zero-fills the K=700 edge,
//            LDS padding via D# pad fields for conflict-free fragment loads).
//   Phase 2: 200-step LIF recurrence, 16 batch rows per persistent block,
//            spikes in LDS (WMMA A layout), syn/mem in registers in the C/D
//            layout, w_hidden bf16-packed and L2-resident. Fragment loads are
//            software-pipelined so WMMAs don't serialize on loadcnt 0.
// ---------------------------------------------------------------------------

typedef __bf16 v16bf __attribute__((ext_vector_type(16)));
typedef __bf16 v8bf  __attribute__((ext_vector_type(8)));
typedef float  v8f   __attribute__((ext_vector_type(8)));
typedef unsigned int v4u __attribute__((ext_vector_type(4)));
typedef int    v8i   __attribute__((ext_vector_type(8)));
typedef int    v4i   __attribute__((ext_vector_type(4)));

#define T_STEPS 200
#define NIN     700
#define NH      1024
#define MROWS   (256 * T_STEPS)            /* 51200 flattened (b,t) rows */
#define ALPHA_C 0.9048374180359595f
#define BETA_C  0.9512294245007141f

static __device__ __forceinline__ __bf16 f2bf(float f) {
  unsigned u = __builtin_bit_cast(unsigned, f);
  unsigned r = u + 0x7FFFu + ((u >> 16) & 1u);          // round-to-nearest-even
  return __builtin_bit_cast(__bf16, (unsigned short)(r >> 16));
}

// ---------------------------------------------------------------------------
// TDM: async 2D f32 tile load Global -> LDS (6-arg builtin on this toolchain).
// D# group1 word0: data_size=4B (code 2), pad_enable=1,
// pad_interval code 4 (pad after every 32 DWORDs), pad_amount code 3 (4 DW)
// => LDS row stride 36 DWORDs (bank-conflict-free b128 fragment loads).
// OOB reads (x >= tdim0) return zero: handles the K=700 boundary in HW.
// ---------------------------------------------------------------------------
static __device__ __forceinline__ void tdm_load_2d_f32(
    unsigned lds_addr, const float* gptr, unsigned tdim0, unsigned tdim1,
    unsigned tstride0, unsigned tile0, unsigned tile1) {
  unsigned long long ga = (unsigned long long)(const void*)gptr;
  v4u g0;
  g0[0] = 1u;                                           // count=1 (user D#)
  g0[1] = lds_addr;                                     // LDS byte address
  g0[2] = (unsigned)ga;                                 // global_addr lo
  g0[3] = (unsigned)((ga >> 32) & 0x01FFFFFFu) | (2u << 30); // addr hi | type=2
  v8i g1;
  g1[0] = (int)((2u << 16) | (1u << 20) | (4u << 22) | (3u << 25));
  g1[1] = (int)((tdim0 & 0xFFFFu) << 16);               // tensor_dim0 lo16
  g1[2] = (int)((tdim0 >> 16) | ((tdim1 & 0xFFFFu) << 16));
  g1[3] = (int)((tdim1 >> 16) | (tile0 << 16));         // tile_dim0
  g1[4] = (int)(tile1 & 0xFFFFu);                       // tile_dim1 (tile_dim2=0)
  g1[5] = (int)tstride0;                                // tensor_dim0_stride lo32
  g1[6] = 0;
  g1[7] = 0;
  v4i gz4 = {0, 0, 0, 0};                               // 2D: groups 2/3 unused
  v8i gz8 = {0, 0, 0, 0, 0, 0, 0, 0};
  __builtin_amdgcn_tensor_load_to_lds(g0, g1, gz4, gz4, gz8, 0);
}

// ---------------------------------------------------------------------------
// Pack W[K][1024] (f32, row-major) -> bf16 WMMA B-fragment layout.
// Tile (kt,nt): lane l holds column n = nt*16+(l&15), K runs of 8 at
// kt*32+(l>>4)*8 and kt*32+16+(l>>4)*8 (32 B per lane). Zero-pads K up to Kp.
// ---------------------------------------------------------------------------
__global__ void pack_w(const float* __restrict__ W, int Kreal, int numTiles,
                       __bf16* __restrict__ dst) {
  int tid = blockIdx.x * blockDim.x + threadIdx.x;
  if (tid >= numTiles * 32) return;
  int lane = tid & 31;
  int tile = tid >> 5;
  int kt = tile >> 6;
  int nt = tile & 63;
  int k0 = kt * 32 + (lane >> 4) * 8;
  int n  = nt * 16 + (lane & 15);
  v16bf v;
#pragma unroll
  for (int i = 0; i < 16; ++i) {
    int k = k0 + i + ((i >> 3) << 3);
    float x = (k < Kreal) ? W[(size_t)k * NH + n] : 0.0f;
    v[i] = f2bf(x);
  }
  *(v16bf*)(dst + (size_t)tid * 16) = v;
}

// ---------------------------------------------------------------------------
// Phase 1: h1[51200][1024] = inputs[51200][700] @ w_input.
// Block tile 64(M) x 128(N), 8 waves in 4x2; wave owns 16x64 (4 accums).
// A tiles arrive via TDM into a double-buffered padded LDS image.
// ---------------------------------------------------------------------------
__global__ void __launch_bounds__(256)
input_gemm(const float* __restrict__ inp, const __bf16* __restrict__ wpk,
           float* __restrict__ h1) {
  __shared__ float Af[2][64][36];                       // 36 DW row stride (TDM pad)
  const int nblock = blockIdx.x;                        // 0..7
  const int mblock = blockIdx.y;                        // 0..799
  const int tid = threadIdx.x, lane = tid & 31, wave = tid >> 5;
  const int hl = lane >> 4, ll = lane & 15;
  const int mrow = (wave >> 1) * 16;
  const int nc0  = (wave & 1) * 64;

  v8f acc[4];
#pragma unroll
  for (int j = 0; j < 4; ++j)
#pragma unroll
    for (int i = 0; i < 8; ++i) acc[j][i] = 0.0f;

  const float* gbase = inp + (size_t)mblock * 64 * NIN;

  if (wave == 0) {
    tdm_load_2d_f32((unsigned)(unsigned long long)&Af[0][0][0], gbase,
                    NIN, 65536u, NIN, 32u, 64u);
    __builtin_amdgcn_s_wait_tensorcnt(0);
  }
  __syncthreads();

  // B fragments: base + (kc*65536 + j*1024) byte immediates after unroll.
  const v16bf* bw = (const v16bf*)wpk
                  + (size_t)(nblock * 8 + (wave & 1) * 4) * 32 + lane;

  for (int kc = 0; kc < 22; ++kc) {                     // Kp = 704
    if (wave == 0 && kc + 1 < 22) {                     // prefetch next tile
      tdm_load_2d_f32((unsigned)(unsigned long long)&Af[(kc + 1) & 1][0][0],
                      gbase + (kc + 1) * 32,
                      (unsigned)(NIN - (kc + 1) * 32),  // HW zero-fills K edge
                      65536u, NIN, 32u, 64u);
    }
    // Issue all four B fragments first: one clause of 8 b128 loads; the
    // first WMMA can then proceed with later loads still in flight.
    v16bf b0 = bw[(size_t)kc * 2048 +  0];
    v16bf b1 = bw[(size_t)kc * 2048 + 32];
    v16bf b2 = bw[(size_t)kc * 2048 + 64];
    v16bf b3 = bw[(size_t)kc * 2048 + 96];
    // A fragment (16x32 bf16): lanes<16 K 0..7 & 16..23; lanes>=16 K 8..15 & 24..31
    const float* arow = &Af[kc & 1][mrow + ll][hl * 8];
    v8f flo = *(const v8f*)arow;
    v8f fhi = *(const v8f*)(arow + 16);
    v16bf a;
#pragma unroll
    for (int i = 0; i < 8; ++i) {
      a[i]     = (__bf16)flo[i];
      a[8 + i] = (__bf16)fhi[i];
    }
    acc[0] = __builtin_amdgcn_wmma_f32_16x16x32_bf16(false, a, false, b0,
                                                     (short)0, acc[0], false, false);
    acc[1] = __builtin_amdgcn_wmma_f32_16x16x32_bf16(false, a, false, b1,
                                                     (short)0, acc[1], false, false);
    acc[2] = __builtin_amdgcn_wmma_f32_16x16x32_bf16(false, a, false, b2,
                                                     (short)0, acc[2], false, false);
    acc[3] = __builtin_amdgcn_wmma_f32_16x16x32_bf16(false, a, false, b3,
                                                     (short)0, acc[3], false, false);
    if (wave == 0) __builtin_amdgcn_s_wait_tensorcnt(0);
    __syncthreads();
  }

#pragma unroll
  for (int j = 0; j < 4; ++j)
#pragma unroll
    for (int r = 0; r < 8; ++r) {
      size_t row = (size_t)(mblock * 64 + mrow + r + 8 * hl);
      int col = nblock * 128 + nc0 + j * 16 + ll;
      h1[row * NH + col] = acc[j][r];
    }
}

// ---------------------------------------------------------------------------
// Phase 2: persistent recurrence. 16 blocks x 1024 threads (32 waves).
// Block owns 16 batch rows; wave owns a 32-column strip (2 WMMA N-tiles).
// B/A fragment loads run one K-chunk ahead of the WMMAs (distinct registers
// so the backend can use partial loadcnt waits).
// ---------------------------------------------------------------------------
__global__ void __launch_bounds__(1024)
recurrent(const __bf16* __restrict__ whpk, float* __restrict__ out) {
  __shared__ __bf16 spk[16][1032];                      // 2064 B row stride
  const int tid = threadIdx.x, lane = tid & 31, wave = tid >> 5;
  const int hl = lane >> 4, ll = lane & 15;
  const int bbase = blockIdx.x * 16;
  const int wcol  = wave * 32;

  float syn[2][8], mem[2][8];
#pragma unroll
  for (int j = 0; j < 2; ++j)
#pragma unroll
    for (int r = 0; r < 8; ++r) { syn[j][r] = 0.0f; mem[j][r] = 0.0f; }

  for (int i = tid; i < 16 * 1032; i += 1024)
    (&spk[0][0])[i] = __builtin_bit_cast(__bf16, (unsigned short)0);
  __syncthreads();

  // Invariant base pointers; everything below is base + immediate.
  const v16bf* bw = (const v16bf*)whpk + (size_t)(wave * 2) * 32 + lane;
  const __bf16* ap = &spk[ll][hl * 8];
  __bf16* sout = &spk[8 * hl][wcol + ll];

  const size_t base0 = (size_t)(bbase + 8 * hl) * T_STEPS * NH + wcol + ll;
  float* sp = out + base0;                              // spk_rec stream
  float* hp = out + (size_t)MROWS * NH + base0;         // h1 in / mem_rec out

  for (int t = 0; t < T_STEPS; ++t) {
    v8f acc[2];
#pragma unroll
    for (int j = 0; j < 2; ++j)
#pragma unroll
      for (int r = 0; r < 8; ++r)
        acc[j][r] = hp[(size_t)r * (T_STEPS * NH) + j * 16];   // h_t drive

    // Software-pipelined out(t-1) @ W: prologue loads chunk 0.
    v16bf b0 = bw[0];
    v16bf b1 = bw[32];
    v8bf lo = *(const v8bf*)(ap);
    v8bf hi = *(const v8bf*)(ap + 16);
#pragma unroll
    for (int kt = 0; kt < 32; ++kt) {
      // Issue next chunk's loads into fresh registers ((kt+1)&31 keeps the
      // epilogue branch-free; the wrapped loads are redundant but harmless).
      const int nk = (kt + 1) & 31;
      v16bf nb0 = bw[(size_t)nk * 2048];
      v16bf nb1 = bw[(size_t)nk * 2048 + 32];
      v8bf nlo = *(const v8bf*)(ap + nk * 32);
      v8bf nhi = *(const v8bf*)(ap + nk * 32 + 16);
      v16bf a = __builtin_shufflevector(lo, hi, 0, 1, 2, 3, 4, 5, 6, 7,
                                        8, 9, 10, 11, 12, 13, 14, 15);
      acc[0] = __builtin_amdgcn_wmma_f32_16x16x32_bf16(false, a, false, b0,
                                                       (short)0, acc[0], false, false);
      acc[1] = __builtin_amdgcn_wmma_f32_16x16x32_bf16(false, a, false, b1,
                                                       (short)0, acc[1], false, false);
      b0 = nb0; b1 = nb1; lo = nlo; hi = nhi;
    }
    __syncthreads();                                    // all waves read spk(t-1)

#pragma unroll
    for (int j = 0; j < 2; ++j)
#pragma unroll
      for (int r = 0; r < 8; ++r) {
        float m = mem[j][r], s = syn[j][r];
        float no = (m > 1.0f) ? 1.0f : 0.0f;            // spike(mem - 1)
        size_t off = (size_t)r * (T_STEPS * NH) + j * 16;
        sp[off] = no;                                   // spk_rec(t)
        hp[off] = m;                                    // mem_rec(t) = pre-update
        syn[j][r] = ALPHA_C * s + acc[j][r];
        mem[j][r] = (BETA_C * m + s) * (1.0f - no);
        sout[r * 1032 + j * 16] = f2bf(no);             // A matrix for t+1
      }
    if (t + 1 < T_STEPS)
      __builtin_prefetch(hp + NH, 0, 1);                // warm next h_t line
    __syncthreads();                                    // spk(t) visible
    sp += NH; hp += NH;
  }
}

extern "C" void kernel_launch(void* const* d_in, const int* in_sizes, int n_in,
                              void* d_out, int out_size, void* d_ws, size_t ws_size,
                              hipStream_t stream) {
  (void)in_sizes; (void)n_in; (void)out_size; (void)ws_size;
  const float* inputs   = (const float*)d_in[0];   // [256,200,700]
  const float* w_input  = (const float*)d_in[1];   // [700,1024]
  const float* w_hidden = (const float*)d_in[2];   // [1024,1024]
  float* out = (float*)d_out;                      // [spk_rec | mem_rec]

  __bf16* wh_pk = (__bf16*)d_ws;                   // 2 MB packed w_hidden
  __bf16* wi_pk = wh_pk + (size_t)32 * 64 * 512;   // 1.41 MB packed w_input

  {
    int tiles = 32 * 64;                           // Kp=1024
    pack_w<<<(tiles * 32 + 255) / 256, 256, 0, stream>>>(w_hidden, NH, tiles, wh_pk);
  }
  {
    int tiles = 22 * 64;                           // Kp=704
    pack_w<<<(tiles * 32 + 255) / 256, 256, 0, stream>>>(w_input, NIN, tiles, wi_pk);
  }
  input_gemm<<<dim3(8, 800), 256, 0, stream>>>(inputs, wi_pk, out + (size_t)MROWS * NH);
  recurrent<<<16, 1024, 0, stream>>>(wh_pk, out);
}
